// SelfAttention_15599321219352
// MI455X (gfx1250) — matrix operand order repaired and checked by reading, not compiled, not run on hardware
//
#include <hip/hip_runtime.h>
#include <hip/hip_bf16.h>
#include <cstdint>
#include <cstddef>

// Problem shape (fixed by the reference).
#define BATCH   8
#define CCH     512
#define HW      4096
#define THREE_C 1536

// LDS tile row pitch in bf16 elements: 32 data + 8 pad = 80 bytes.
// Fragment reads hit 16 distinct banks across the 16 lanes (row*20 mod 64).
#define TPITCH  40

typedef __attribute__((ext_vector_type(16))) __bf16        v16bf;
typedef __attribute__((ext_vector_type(8)))  float          v8f;
typedef __attribute__((ext_vector_type(4)))  unsigned int   u32x4;
typedef __attribute__((ext_vector_type(8)))  unsigned int   u32x8;
typedef __attribute__((ext_vector_type(8)))  unsigned short u16x8;

// fp32 -> bf16 round-to-nearest-even
__device__ __forceinline__ unsigned short f2bf(float f) {
  unsigned int u = __builtin_bit_cast(unsigned int, f);
  u += 0x7FFFu + ((u >> 16) & 1u);
  return (unsigned short)(u >> 16);
}

// ---------------------------------------------------------------------------
// gfx1250 async global->LDS copy, GVS mode:
//   mem_addr = SGPR_u64(base) + VGPR_i32(goff);  LDS[vdst] = 16 bytes
// vdst is the wave-relative LDS byte address (low 32 bits of a generic
// pointer to __shared__, per the aperture mapping).
// Tracked by ASYNCcnt.
// ---------------------------------------------------------------------------
__device__ __forceinline__ void async_copy_b128(unsigned lds_addr, unsigned goff,
                                                const void* base) {
  unsigned long long b = (unsigned long long)(uintptr_t)base;
  asm volatile("global_load_async_to_lds_b128 %0, %1, %2"
               :: "v"(lds_addr), "v"(goff), "s"(b)
               : "memory");
}

__device__ __forceinline__ void wait_async_le3() {
  asm volatile("s_wait_asynccnt 0x3" ::: "memory");
}
__device__ __forceinline__ void wait_async_0() {
  asm volatile("s_wait_asynccnt 0x0" ::: "memory");
}

// ---------------------------------------------------------------------------
// WMMA fragment loaders from the padded LDS tiles (wave32 layouts for
// v_wmma_f32_16x16x32_bf16).
//
// A (16x32): lane L<16 : row M=L,    K={0..7,16..23}
//            lane L>=16: row M=L-16, K={8..15,24..31}
// B (32x16): lane L<16 : col N=L,    K={0..7,16..23} ; lane L>=16: col N=L-16, K={8..15,24..31}
// C/D: element r of v8f = (M = r + 8*(lane>>4), N = lane&15).
// ---------------------------------------------------------------------------
__device__ __forceinline__ v16bf lds_frag_a(const unsigned short* base,
                                            int row0, int lane) {
  const unsigned short* p = base + (row0 + (lane & 15)) * TPITCH + ((lane >> 4) << 3);
  u32x4 lo = *reinterpret_cast<const u32x4*>(p);       // K = kb .. kb+7
  u32x4 hi = *reinterpret_cast<const u32x4*>(p + 16);  // K = 16+kb .. 16+kb+7
  u32x8 u;
  u[0] = lo[0]; u[1] = lo[1]; u[2] = lo[2]; u[3] = lo[3];
  u[4] = hi[0]; u[5] = hi[1]; u[6] = hi[2]; u[7] = hi[3];
  return __builtin_bit_cast(v16bf, u);
}

__device__ __forceinline__ v16bf lds_frag_b(const unsigned short* base,
                                            int row0, int lane) {
  const unsigned short* p = base + (row0 + (lane & 15)) * TPITCH + ((lane >> 4) << 3);
  u32x4 lo = *reinterpret_cast<const u32x4*>(p);       // K = kb .. kb+7
  u32x4 hi = *reinterpret_cast<const u32x4*>(p + 16);  // K = 16+kb .. 16+kb+7
  u32x8 u;
  u[0] = lo[0]; u[1] = lo[1]; u[2] = lo[2]; u[3] = lo[3];
  u[4] = hi[0]; u[5] = hi[1]; u[6] = hi[2]; u[7] = hi[3];
  return __builtin_bit_cast(v16bf, u);
}

__device__ __forceinline__ v8f wmma_bf16(v16bf a, v16bf b, v8f c) {
  return __builtin_amdgcn_wmma_f32_16x16x32_bf16(
      /*neg_a=*/false, a, /*neg_b=*/false, b,
      /*c_mod=*/(short)0, c, /*reuse_a=*/false, /*reuse_b=*/false);
}

// ---------------------------------------------------------------------------
// Common GEMM core: D(64M x 128N per block) = A(row-major, lda) x B where
// column n of B is row n of Bt (N-major, ldb).  K multiple of 32.
// 8 waves/block, wave tile 32x32 (2x2 WMMA).  A/B tiles staged to LDS with
// double-buffered async copies.
// ---------------------------------------------------------------------------
template <typename Epi>
__device__ __forceinline__ void gemm_core(const unsigned short* __restrict__ A,
                                          int lda,
                                          const unsigned short* __restrict__ Bt,
                                          int ldb, int K, Epi&& epi) {
  __shared__ __align__(16) unsigned short As[2][64 * TPITCH];
  __shared__ __align__(16) unsigned short Bs[2][128 * TPITCH];

  const int tid  = threadIdx.x;
  const int lane = tid & 31;
  const int wv   = tid >> 5;
  const int m_blk = blockIdx.y * 64;
  const int n_blk = blockIdx.x * 128;

  // Staging assignment: 256 threads cover 64x(4x16B) A segments and
  // 128x(4x16B) B segments (2 rows per thread for B).
  const int row  = tid >> 2;   // 0..63
  const int seg  = tid & 3;    // 0..3, 16B each
  const unsigned a_goff  = (unsigned)(m_blk + row) * (unsigned)lda * 2u + seg * 16u;
  const unsigned b_goff0 = (unsigned)(n_blk + row) * (unsigned)ldb * 2u + seg * 16u;
  const unsigned b_goff1 = (unsigned)(n_blk + 64 + row) * (unsigned)ldb * 2u + seg * 16u;

  unsigned a_lds[2], b_lds0[2], b_lds1[2];
  #pragma unroll
  for (int i = 0; i < 2; ++i) {
    a_lds[i]  = (unsigned)(uintptr_t)&As[i][row * TPITCH + seg * 8];
    b_lds0[i] = (unsigned)(uintptr_t)&Bs[i][row * TPITCH + seg * 8];
    b_lds1[i] = (unsigned)(uintptr_t)&Bs[i][(64 + row) * TPITCH + seg * 8];
  }

  auto issue = [&](int buf, int k0) {
    unsigned kb = (unsigned)k0 * 2u;
    async_copy_b128(a_lds[buf],  a_goff  + kb, A);
    async_copy_b128(b_lds0[buf], b_goff0 + kb, Bt);
    async_copy_b128(b_lds1[buf], b_goff1 + kb, Bt);
  };

  const int m_lo = (wv & 1) * 32;
  const int n_lo = (wv >> 1) * 32;

  v8f acc00 = {}, acc01 = {}, acc10 = {}, acc11 = {};
  issue(0, 0);
  int buf = 0;
  for (int k0 = 0; k0 < K; k0 += 32) {
    const bool more = (k0 + 32 < K);
    if (more) issue(buf ^ 1, k0 + 32);       // prefetch next stage
    if (more) wait_async_le3();              // own stage-i loads complete
    else      wait_async_0();
    __syncthreads();                         // all waves' staging visible

    const unsigned short* as = As[buf];
    const unsigned short* bs = Bs[buf];
    v16bf a0 = lds_frag_a(as, m_lo,      lane);
    v16bf a1 = lds_frag_a(as, m_lo + 16, lane);
    v16bf b0 = lds_frag_b(bs, n_lo,      lane);
    v16bf b1 = lds_frag_b(bs, n_lo + 16, lane);
    acc00 = wmma_bf16(a0, b0, acc00);
    acc01 = wmma_bf16(a0, b1, acc01);
    acc10 = wmma_bf16(a1, b0, acc10);
    acc11 = wmma_bf16(a1, b1, acc11);

    __syncthreads();                         // done reading buf before overwrite
    buf ^= 1;
  }

  v8f accs[4] = {acc00, acc01, acc10, acc11};
  epi(accs, m_blk + m_lo, n_blk + n_lo, lane);
}

// ---------------------------------------------------------------------------
// Kernel 0: convert fp32 weights to bf16 (qkv_w: 1536x512, out_w: 512x512)
// ---------------------------------------------------------------------------
__global__ void wconv_kernel(const float* __restrict__ qkv_w,
                             const float* __restrict__ out_w,
                             unsigned short* __restrict__ qkv_wbf,
                             unsigned short* __restrict__ out_wbf) {
  int i = blockIdx.x * 256 + threadIdx.x;
  if (i < THREE_C * CCH) qkv_wbf[i] = f2bf(qkv_w[i]);
  if (i < CCH * CCH)     out_wbf[i] = f2bf(out_w[i]);
}

// ---------------------------------------------------------------------------
// Kernel 1: GroupNorm(32 groups) and transpose to xnT [b, hw, c] bf16.
// ---------------------------------------------------------------------------
__global__ void groupnorm_kernel(const float* __restrict__ x,
                                 const float* __restrict__ gw,
                                 const float* __restrict__ gb,
                                 unsigned short* __restrict__ xnT) {
  const int tid = threadIdx.x;
  const int b = blockIdx.x >> 5;
  const int g = blockIdx.x & 31;
  const float* xp = x + ((size_t)b * CCH + g * 16) * HW;

  float s = 0.f, ss = 0.f;
  for (int i = tid; i < 16 * HW; i += 256) {
    float v = xp[i];
    s += v; ss += v * v;
  }
  __shared__ float rs[256], rss[256];
  rs[tid] = s; rss[tid] = ss;
  __syncthreads();
  for (int o = 128; o > 0; o >>= 1) {
    if (tid < o) { rs[tid] += rs[tid + o]; rss[tid] += rss[tid + o]; }
    __syncthreads();
  }
  const float inv_n = 1.0f / (16.0f * (float)HW);
  float mean = rs[0] * inv_n;
  float var  = fmaxf(rss[0] * inv_n - mean * mean, 0.0f);
  float rstd = rsqrtf(var + 1e-5f);

  for (int i = tid; i < 16 * HW; i += 256) {
    int chL = i >> 12;
    int n   = i & (HW - 1);
    int ch  = g * 16 + chL;
    float v = (xp[i] - mean) * rstd * gw[ch] + gb[ch];
    xnT[((size_t)b * HW + n) * CCH + ch] = f2bf(v);
  }
}

// ---------------------------------------------------------------------------
// Kernel 2: QKV GEMM.  M=1536, N=4096, K=512.
// ---------------------------------------------------------------------------
__global__ void gemm_qkv_kernel(const unsigned short* __restrict__ Wbf,
                                const unsigned short* __restrict__ xnT,
                                const float* __restrict__ bias,
                                unsigned short* __restrict__ qb,
                                unsigned short* __restrict__ kbuf,
                                unsigned short* __restrict__ vT) {
  const int b = blockIdx.z;
  const unsigned short* Bt = xnT + (size_t)b * HW * CCH;
  gemm_core(Wbf, CCH, Bt, CCH, CCH,
            [&](const v8f* accs, int m_base, int n_base, int lane) {
    const int half = (lane >> 4) << 3;
    #pragma unroll
    for (int t = 0; t < 4; ++t) {
      const v8f acc = accs[t];
      const int mt = m_base + ((t >> 1) << 4);
      const int nt = n_base + ((t & 1) << 4);
      const int n = nt + (lane & 15);
      #pragma unroll
      for (int r = 0; r < 8; ++r) {
        int d = mt + half + r;
        unsigned short bv = f2bf(acc[r] + bias[d]);
        if (d < CCH)
          qb[((size_t)b * CCH + d) * HW + n] = bv;
        else if (d < 2 * CCH)
          kbuf[((size_t)b * CCH + (d - CCH)) * HW + n] = bv;
        else
          vT[((size_t)b * HW + n) * CCH + (d - 2 * CCH)] = bv;
      }
    }
  });
}

// ---------------------------------------------------------------------------
// Kernel 3: scores[c,d] = sum_n q[c,n]*k[d,n].  M=N=512, K=4096.
// ---------------------------------------------------------------------------
__global__ void gemm_scores_kernel(const unsigned short* __restrict__ qb,
                                   const unsigned short* __restrict__ kbuf,
                                   float* __restrict__ scores) {
  const int b = blockIdx.z;
  const unsigned short* A  = qb   + (size_t)b * CCH * HW;
  const unsigned short* Bt = kbuf + (size_t)b * CCH * HW;
  float* S = scores + (size_t)b * CCH * CCH;
  gemm_core(A, HW, Bt, HW, HW,
            [&](const v8f* accs, int m_base, int n_base, int lane) {
    const int half = (lane >> 4) << 3;
    #pragma unroll
    for (int t = 0; t < 4; ++t) {
      const v8f acc = accs[t];
      const int mt = m_base + ((t >> 1) << 4);
      const int nt = n_base + ((t & 1) << 4);
      const int n = nt + (lane & 15);
      #pragma unroll
      for (int r = 0; r < 8; ++r)
        S[(size_t)(mt + half + r) * CCH + n] = acc[r];
    }
  });
}

// ---------------------------------------------------------------------------
// Kernel 4: row softmax over 512 (fp32 in, bf16 out, attn [b,c,c] row-major)
// ---------------------------------------------------------------------------
__global__ void softmax_kernel(const float* __restrict__ scores,
                               unsigned short* __restrict__ attn) {
  const int tid = threadIdx.x;
  const size_t row = blockIdx.x;
  const float* s = scores + row * CCH;

  float v0 = s[tid], v1 = s[tid + 256];
  __shared__ float red[256];
  red[tid] = fmaxf(v0, v1);
  __syncthreads();
  for (int o = 128; o > 0; o >>= 1) {
    if (tid < o) red[tid] = fmaxf(red[tid], red[tid + o]);
    __syncthreads();
  }
  float mx = red[0];
  __syncthreads();
  float e0 = __expf(v0 - mx), e1 = __expf(v1 - mx);
  red[tid] = e0 + e1;
  __syncthreads();
  for (int o = 128; o > 0; o >>= 1) {
    if (tid < o) red[tid] += red[tid + o];
    __syncthreads();
  }
  float inv = 1.0f / red[0];
  attn[row * CCH + tid]       = f2bf(e0 * inv);
  attn[row * CCH + tid + 256] = f2bf(e1 * inv);
}

// ---------------------------------------------------------------------------
// Kernel 5: out = attn @ v.  M=512, N=4096, K=512.  Writes outT [b,hw,c]
// with 16B vector stores (lane's 8 D elements are M-contiguous).
// ---------------------------------------------------------------------------
__global__ void gemm_av_kernel(const unsigned short* __restrict__ attn,
                               const unsigned short* __restrict__ vT,
                               unsigned short* __restrict__ outT) {
  const int b = blockIdx.z;
  const unsigned short* A  = attn + (size_t)b * CCH * CCH;
  const unsigned short* Bt = vT   + (size_t)b * HW * CCH;
  unsigned short* O = outT + (size_t)b * HW * CCH;
  gemm_core(A, CCH, Bt, CCH, CCH,
            [&](const v8f* accs, int m_base, int n_base, int lane) {
    const int half = (lane >> 4) << 3;
    #pragma unroll
    for (int t = 0; t < 4; ++t) {
      const v8f acc = accs[t];
      const int mt = m_base + ((t >> 1) << 4);
      const int nt = n_base + ((t & 1) << 4);
      const int n = nt + (lane & 15);
      u16x8 pack;
      #pragma unroll
      for (int r = 0; r < 8; ++r) pack[r] = f2bf(acc[r]);
      *reinterpret_cast<u16x8*>(O + (size_t)n * CCH + mt + half) = pack;
    }
  });
}

// ---------------------------------------------------------------------------
// Kernel 6: proj = out_w @ out + out_b + x (residual), fp32 output.
// ---------------------------------------------------------------------------
__global__ void gemm_proj_kernel(const unsigned short* __restrict__ Wbf,
                                 const unsigned short* __restrict__ outT,
                                 const float* __restrict__ x,
                                 const float* __restrict__ bias,
                                 float* __restrict__ out) {
  const int b = blockIdx.z;
  const unsigned short* Bt = outT + (size_t)b * HW * CCH;
  gemm_core(Wbf, CCH, Bt, CCH, CCH,
            [&](const v8f* accs, int m_base, int n_base, int lane) {
    const int half = (lane >> 4) << 3;
    #pragma unroll
    for (int t = 0; t < 4; ++t) {
      const v8f acc = accs[t];
      const int mt = m_base + ((t >> 1) << 4);
      const int nt = n_base + ((t & 1) << 4);
      const int n = nt + (lane & 15);
      #pragma unroll
      for (int r = 0; r < 8; ++r) {
        int d = mt + half + r;
        size_t idx = ((size_t)b * CCH + d) * HW + n;
        out[idx] = x[idx] + acc[r] + bias[d];
      }
    }
  });
}

// ---------------------------------------------------------------------------
extern "C" void kernel_launch(void* const* d_in, const int* in_sizes, int n_in,
                              void* d_out, int out_size, void* d_ws, size_t ws_size,
                              hipStream_t stream) {
  const float* x      = (const float*)d_in[0];
  const float* norm_w = (const float*)d_in[1];
  const float* norm_b = (const float*)d_in[2];
  const float* qkv_w  = (const float*)d_in[3];
  const float* qkv_b  = (const float*)d_in[4];
  const float* out_w  = (const float*)d_in[5];
  const float* out_b  = (const float*)d_in[6];
  float* out = (float*)d_out;

  char* ws = (char*)d_ws;
  size_t off = 0;
  auto take = [&](size_t bytes) -> char* {
    char* p = ws + off;
    off = (off + bytes + 255) & ~(size_t)255;
    return p;
  };

  unsigned short* qkv_wbf = (unsigned short*)take((size_t)THREE_C * CCH * 2);
  unsigned short* out_wbf = (unsigned short*)take((size_t)CCH * CCH * 2);
  unsigned short* xnT     = (unsigned short*)take((size_t)BATCH * HW * CCH * 2);
  unsigned short* qb      = (unsigned short*)take((size_t)BATCH * CCH * HW * 2);
  unsigned short* kbuf    = (unsigned short*)take((size_t)BATCH * CCH * HW * 2);
  unsigned short* vT      = (unsigned short*)take((size_t)BATCH * HW * CCH * 2);
  float*          scores  = (float*)take((size_t)BATCH * CCH * CCH * 4);
  unsigned short* attn    = (unsigned short*)take((size_t)BATCH * CCH * CCH * 2);
  unsigned short* outT    = xnT;  // xnT is dead after the QKV GEMM (stream order)

  wconv_kernel<<<(THREE_C * CCH + 255) / 256, 256, 0, stream>>>(
      qkv_w, out_w, qkv_wbf, out_wbf);
  groupnorm_kernel<<<BATCH * 32, 256, 0, stream>>>(x, norm_w, norm_b, xnT);
  gemm_qkv_kernel<<<dim3(HW / 128, THREE_C / 64, BATCH), 256, 0, stream>>>(
      qkv_wbf, xnT, qkv_b, qb, kbuf, vT);
  gemm_scores_kernel<<<dim3(CCH / 128, CCH / 64, BATCH), 256, 0, stream>>>(
      qb, kbuf, scores);
  softmax_kernel<<<BATCH * CCH, 256, 0, stream>>>(scores, attn);
  gemm_av_kernel<<<dim3(HW / 128, CCH / 64, BATCH), 256, 0, stream>>>(
      attn, vT, outT);
  gemm_proj_kernel<<<dim3(HW / 128, CCH / 64, BATCH), 256, 0, stream>>>(
      out_wbf, outT, x, out_b, out);
}
